// MemoryModule_3994319585528
// MI455X (gfx1250) — matrix operand order repaired and checked by reading, not compiled, not run on hardware
//
#include <hip/hip_runtime.h>
#include <hip/hip_bf16.h>

// ---------------------------------------------------------------------------
// Problem constants (from reference): B=2, T=4, Ck=64, Cv=256, H=W=64
// pool stride 2 -> h=w=32, M = T*h*w = 4096, N = H*W = 4096
// ---------------------------------------------------------------------------
#define B_   2
#define T_   4
#define CK   64
#define CV   256
#define H_   64
#define W_   64
#define N_   4096          // H*W
#define M_   4096          // T*32*32
#define KCH  64            // keys per flash iteration
#define WAVES 4            // waves per workgroup (wave32)
#define ROWS 16            // N-rows per wave (WMMA M dim)
#define NITER (M_ / KCH)   // 64

// scale = Ck^-0.5 * log2(e): softmax computed in log2 domain (exp2f)
#define QSCALE 0.1803368801111204f

typedef __attribute__((ext_vector_type(16))) __bf16 v16bf;
typedef __attribute__((ext_vector_type(8)))  float  v8f;
typedef __attribute__((ext_vector_type(4)))  unsigned int u32x4;
typedef __attribute__((ext_vector_type(8)))  int i32x8;
typedef __attribute__((ext_vector_type(4)))  int i32x4;

#if defined(__has_include)
#  if __has_include(<hip/amd_detail/amd_gfx1250_TDM.h>)
#    define TDM_6ARG 1
#  else
#    define TDM_6ARG 0
#  endif
#else
#  define TDM_6ARG 0
#endif

// ---------------------------------------------------------------------------
// Fragment loaders (k order chosen here for both 16-bit WMMA operands, wave32)
//   A-frag [16 x 32]: lane L -> row = L%16 ; element e -> K = (L/16)*16 + e
//   B-frag [32 x 16]: lane L -> col = L%16 ; element e -> K = (L/16)*16 + e
// ---------------------------------------------------------------------------
__device__ __forceinline__ v16bf load_frag_contig(const __bf16* p) {
    union { v16bf v; uint4 u[2]; } t;
    t.u[0] = *reinterpret_cast<const uint4*>(p);
    t.u[1] = *reinterpret_cast<const uint4*>(p + 8);
    return t.v;
}
__device__ __forceinline__ v16bf load_frag_split(const __bf16* p) {
    union { v16bf v; uint4 u[2]; } t;
    t.u[0] = *reinterpret_cast<const uint4*>(p);
    t.u[1] = *reinterpret_cast<const uint4*>(p + 16);
    return t.v;
}

// v_permlane16_b32 xor-shuffle inside each 16-lane row (pure VALU, no LDS)
__device__ __forceinline__ float permlane16_xor(float v, int lo, int hi) {
    int s = __builtin_bit_cast(int, v);
    int r = __builtin_amdgcn_permlane16(s, s, lo, hi, false, false);
    return __builtin_bit_cast(float, r);
}
// 4-bit select tables: sel[i] = i ^ k, packed little-endian per lane
#define X1_LO 0x67452301
#define X1_HI 0xEFCDAB89
#define X2_LO 0x45670123
#define X2_HI 0xCDEF89AB
#define X4_LO 0x01234567
#define X4_HI 0x89ABCDEF
#define X8_LO 0xFEDCBA98
#define X8_HI 0x76543210

// ---------------------------------------------------------------------------
// Tensor Data Mover: DMA a (tile_d0 x tile_d1) dword tile (row stride
// stride_d0 dwords) from global to LDS (rows packed contiguously in LDS).
// D# layout per CDNA5 ISA; data_size = 4 bytes.
// ---------------------------------------------------------------------------
__device__ __forceinline__ void tdm_load_2d(unsigned lds_addr, const void* gptr,
                                            unsigned tensor_d0, unsigned tensor_d1,
                                            unsigned tile_d0, unsigned tile_d1,
                                            unsigned stride_d0) {
    unsigned long long ga = (unsigned long long)(uintptr_t)gptr;
    u32x4 g0;
    g0.x = 1u;                                              // count=1 (valid user D#)
    g0.y = lds_addr;                                        // LDS byte address
    g0.z = (unsigned)ga;                                    // global_addr[31:0]
    g0.w = (unsigned)((ga >> 32) & 0x01FFFFFFu) | (2u << 30); // addr[56:32] | type=2
    i32x8 g1;
    g1[0] = (int)(2u << 16);                                // data_size = 4B
    g1[1] = (int)((tensor_d0 & 0xFFFFu) << 16);             // tensor_dim0[15:0]
    g1[2] = (int)(((tensor_d0 >> 16) & 0xFFFFu) |
                  ((tensor_d1 & 0xFFFFu) << 16));           // tensor_dim0[31:16] | dim1[15:0]
    g1[3] = (int)(((tensor_d1 >> 16) & 0xFFFFu) |
                  ((tile_d0 & 0xFFFFu) << 16));             // tensor_dim1[31:16] | tile_dim0
    g1[4] = (int)(tile_d1 & 0xFFFFu);                       // tile_dim1 | tile_dim2=0
    g1[5] = (int)stride_d0;                                 // tensor_dim0_stride[31:0]
    g1[6] = 0;
    g1[7] = 0;
    i32x4 z4 = (i32x4)0;
#if TDM_6ARG
    i32x8 z8 = (i32x8)0;
    __builtin_amdgcn_tensor_load_to_lds(g0, g1, z4, z4, z8, 0);
#else
    __builtin_amdgcn_tensor_load_to_lds(g0, g1, z4, z4, 0);
#endif
}

__device__ __forceinline__ unsigned lds_off(const void* p) {
    return (unsigned)(uintptr_t)p;   // LDS offset lives in the low 32 address bits
}

// ---------------------------------------------------------------------------
// Prep kernels: maxpool + layout transform + bf16 convert
// ---------------------------------------------------------------------------
__global__ void pool_keys_kernel(const float* __restrict__ mk, __bf16* __restrict__ Kbf) {
    int idx = blockIdx.x * 256 + threadIdx.x;      // B*M*CK = 1<<19 threads
    int d = idx & 63;
    int m = (idx >> 6) & (M_ - 1);
    int b = idx >> 18;
    int t = m >> 10, rr = m & 1023, y = rr >> 5, x = rr & 31;
    const float* p = mk + ((((size_t)b * T_ + t) * CK + d) * H_ + 2 * y) * W_ + 2 * x;
    float v = fmaxf(fmaxf(p[0], p[1]), fmaxf(p[W_], p[W_ + 1]));
    Kbf[idx] = (__bf16)v;
}

__global__ void pool_vals_kernel(const float* __restrict__ mv, __bf16* __restrict__ Vbf) {
    int idx = blockIdx.x * 256 + threadIdx.x;      // B*CV*M = 1<<21 threads
    int m  = idx & (M_ - 1);
    int cv = (idx >> 12) & (CV - 1);
    int b  = idx >> 20;
    int t = m >> 10, rr = m & 1023, y = rr >> 5, x = rr & 31;
    const float* p = mv + ((((size_t)b * T_ + t) * CV + cv) * H_ + 2 * y) * W_ + 2 * x;
    float v = fmaxf(fmaxf(p[0], p[1]), fmaxf(p[W_], p[W_ + 1]));
    Vbf[idx] = (__bf16)v;
}

__global__ void q_prep_kernel(const float* __restrict__ qk, __bf16* __restrict__ Qbf) {
    int idx = blockIdx.x * 256 + threadIdx.x;      // B*N*CK = 1<<19 threads
    int d = idx & 63;
    int n = (idx >> 6) & (N_ - 1);
    int b = idx >> 18;
    Qbf[idx] = (__bf16)(qk[((size_t)b * CK + d) * N_ + n] * QSCALE);
}

__global__ void copy_qv_kernel(const float* __restrict__ qv, float* __restrict__ out) {
    int idx = blockIdx.x * 256 + threadIdx.x;      // B*CV*N = 1<<21 threads
    int b = idx >> 20;                             // CV*N = 1<<20
    out[(size_t)idx + (size_t)b * (CV * N_)] = qv[idx];
}

// ---------------------------------------------------------------------------
// Flash attention, TDM double-buffered: per wave 16 query rows, 64 keys/iter.
// 40 v_wmma per iteration; K/V chunks DMA'd by the Tensor Data Mover into
// alternating LDS buffers one iteration ahead (TENSORcnt pipelining).
// ---------------------------------------------------------------------------
__global__ __launch_bounds__(WAVES * 32, 1)
void flash_attn_kernel(const __bf16* __restrict__ Qbf,
                       const __bf16* __restrict__ Kbf,
                       const __bf16* __restrict__ Vbf,
                       float* __restrict__ out) {
    __shared__ __align__(16) __bf16 Ks[2][KCH * CK];          // 2 x 8 KB
    __shared__ __align__(16) __bf16 Vs[2][CV * KCH];          // 2 x 32 KB
    __shared__ __align__(16) __bf16 Ps[WAVES][ROWS * KCH];    // 8 KB

    const int tid   = threadIdx.x;
    const int wave  = tid >> 5;
    const int lane  = tid & 31;
    const int lhalf = lane >> 4;
    const int l16   = lane & 15;

    const int ntiles = N_ / (WAVES * ROWS);                   // 64
    const int b  = blockIdx.x / ntiles;
    const int n0 = (blockIdx.x % ntiles) * (WAVES * ROWS) + wave * ROWS;

    const __bf16* gKbase = Kbf + (size_t)b * M_ * CK;
    const __bf16* gVbase = Vbf + (size_t)b * CV * M_;

    // Q A-fragments (scale folded into Qbf)
    const __bf16* qrow = Qbf + (size_t)(b * N_ + n0 + l16) * CK;
    const v16bf q0 = load_frag_contig(qrow + 0 * 32 + lhalf * 16);
    const v16bf q1 = load_frag_contig(qrow + 1 * 32 + lhalf * 16);

    float mrow[8], lrow[8];
#pragma unroll
    for (int r = 0; r < 8; ++r) { mrow[r] = -INFINITY; lrow[r] = 0.0f; }
    v8f acc[16];
#pragma unroll
    for (int c = 0; c < 16; ++c) acc[c] = v8f{};

    // ---- TDM prologue: DMA chunk 0 into buffer 0 ----
    if (wave == 0) {
        tdm_load_2d(lds_off(&Ks[0][0]), gKbase, KCH * CK / 2, 1, KCH * CK / 2, 1, KCH * CK / 2);
        tdm_load_2d(lds_off(&Vs[0][0]), gVbase, M_ / 2, CV, KCH / 2, CV, M_ / 2);
    }

    for (int it = 0; it < NITER; ++it) {
        const int cur = it & 1;

        // ---- issue next chunk's DMA, then wait for current chunk ----
        if (wave == 0) {
            if (it + 1 < NITER) {
                const int m1 = (it + 1) * KCH;
                tdm_load_2d(lds_off(&Ks[cur ^ 1][0]), gKbase + (size_t)m1 * CK,
                            KCH * CK / 2, 1, KCH * CK / 2, 1, KCH * CK / 2);
                tdm_load_2d(lds_off(&Vs[cur ^ 1][0]), gVbase + m1,
                            M_ / 2, CV, KCH / 2, CV, M_ / 2);
                __builtin_amdgcn_s_wait_tensorcnt(2);   // current pair complete
            } else {
                __builtin_amdgcn_s_wait_tensorcnt(0);
            }
        }
        __syncthreads();                                 // publish LDS to all waves

        const __bf16* ks = &Ks[cur][0];
        const __bf16* vs = &Vs[cur][0];

        // ---- S = Q*K^T : four 16x16 key tiles, K-dim 64 split 2x32 ----
        v8f S[4];
#pragma unroll
        for (int j = 0; j < 4; ++j) {
            v16bf bk;
            v8f s = v8f{};
            bk = load_frag_contig(&ks[(j * 16 + l16) * CK + 0 * 32 + lhalf * 16]);
            s = __builtin_amdgcn_wmma_f32_16x16x32_bf16(false, q0, false, bk, (short)0, s, false, false);
            bk = load_frag_contig(&ks[(j * 16 + l16) * CK + 1 * 32 + lhalf * 16]);
            s = __builtin_amdgcn_wmma_f32_16x16x32_bf16(false, q1, false, bk, (short)0, s, false, false);
            S[j] = s;
        }

        // ---- online softmax over this 64-key chunk (log2 domain) ----
        float t[8], alpha[8], sum[8];
#pragma unroll
        for (int r = 0; r < 8; ++r)
            t[r] = fmaxf(fmaxf(S[0][r], S[1][r]), fmaxf(S[2][r], S[3][r]));
#pragma unroll
        for (int r = 0; r < 8; ++r) t[r] = fmaxf(t[r], permlane16_xor(t[r], X1_LO, X1_HI));
#pragma unroll
        for (int r = 0; r < 8; ++r) t[r] = fmaxf(t[r], permlane16_xor(t[r], X2_LO, X2_HI));
#pragma unroll
        for (int r = 0; r < 8; ++r) t[r] = fmaxf(t[r], permlane16_xor(t[r], X4_LO, X4_HI));
#pragma unroll
        for (int r = 0; r < 8; ++r) t[r] = fmaxf(t[r], permlane16_xor(t[r], X8_LO, X8_HI));

        __bf16* pw = &Ps[wave][0];
#pragma unroll
        for (int r = 0; r < 8; ++r) {
            float nm = fmaxf(mrow[r], t[r]);
            alpha[r] = exp2f(mrow[r] - nm);
            mrow[r]  = nm;
            const int row = r + 8 * lhalf;
            float s = 0.0f;
#pragma unroll
            for (int j = 0; j < 4; ++j) {               // store P as it is produced
                float pj = exp2f(S[j][r] - nm);
                pw[row * KCH + j * 16 + l16] = (__bf16)pj;
                s += pj;
            }
            sum[r] = s;
        }
#pragma unroll
        for (int r = 0; r < 8; ++r) sum[r] += permlane16_xor(sum[r], X1_LO, X1_HI);
#pragma unroll
        for (int r = 0; r < 8; ++r) sum[r] += permlane16_xor(sum[r], X2_LO, X2_HI);
#pragma unroll
        for (int r = 0; r < 8; ++r) sum[r] += permlane16_xor(sum[r], X4_LO, X4_HI);
#pragma unroll
        for (int r = 0; r < 8; ++r) sum[r] += permlane16_xor(sum[r], X8_LO, X8_HI);
#pragma unroll
        for (int r = 0; r < 8; ++r) lrow[r] = lrow[r] * alpha[r] + sum[r];

#pragma unroll
        for (int c = 0; c < 16; ++c)
#pragma unroll
            for (int r = 0; r < 8; ++r) acc[c][r] *= alpha[r];

        asm volatile("s_wait_dscnt 0" ::: "memory");    // P stores visible to our loads
        const v16bf pf0 = load_frag_contig(pw + l16 * KCH + 0  + lhalf * 16);
        const v16bf pf1 = load_frag_contig(pw + l16 * KCH + 32 + lhalf * 16);

        // ---- O += P * V : 16 cv tiles x 2 key-chunks ----
#pragma unroll
        for (int c = 0; c < 16; ++c) {
            v16bf bv0 = load_frag_contig(&vs[(c * 16 + l16) * KCH + 0  + lhalf * 16]);
            acc[c] = __builtin_amdgcn_wmma_f32_16x16x32_bf16(false, pf0, false, bv0, (short)0, acc[c], false, false);
            v16bf bv1 = load_frag_contig(&vs[(c * 16 + l16) * KCH + 32 + lhalf * 16]);
            acc[c] = __builtin_amdgcn_wmma_f32_16x16x32_bf16(false, pf1, false, bv1, (short)0, acc[c], false, false);
        }
        __syncthreads();   // all reads of buf[cur] done before it is DMA'd again
    }

    // ---- normalize and write memory half: out[b][CV + cv][n] ----
    float inv[8];
#pragma unroll
    for (int r = 0; r < 8; ++r) inv[r] = 1.0f / lrow[r];
    float* obase = out + ((size_t)b * 2 * CV + CV) * N_;
#pragma unroll
    for (int c = 0; c < 16; ++c) {
        int cv = c * 16 + l16;
#pragma unroll
        for (int r = 0; r < 8; ++r) {
            int n = n0 + r + 8 * lhalf;
            obase[(size_t)cv * N_ + n] = acc[c][r] * inv[r];
        }
    }
}

// ---------------------------------------------------------------------------
extern "C" void kernel_launch(void* const* d_in, const int* in_sizes, int n_in,
                              void* d_out, int out_size, void* d_ws, size_t ws_size,
                              hipStream_t stream) {
    const float* memory_keys   = (const float*)d_in[0];  // [2,4,64,64,64]
    const float* memory_values = (const float*)d_in[1];  // [2,4,256,64,64]
    const float* query_key     = (const float*)d_in[2];  // [2,64,64,64]
    const float* query_value   = (const float*)d_in[3];  // [2,256,64,64]
    float* out = (float*)d_out;                          // [2,512,64,64]

    char* ws = (char*)d_ws;
    __bf16* Kbf = (__bf16*)(ws);                         // B*M*CK  bf16 = 1 MB
    __bf16* Vbf = (__bf16*)(ws + (1u << 20));            // B*CV*M  bf16 = 4 MB
    __bf16* Qbf = (__bf16*)(ws + (5u << 20));            // B*N*CK  bf16 = 1 MB

    pool_keys_kernel<<<(B_ * M_ * CK) / 256, 256, 0, stream>>>(memory_keys, Kbf);
    pool_vals_kernel<<<(B_ * CV * M_) / 256, 256, 0, stream>>>(memory_values, Vbf);
    q_prep_kernel  <<<(B_ * N_ * CK) / 256, 256, 0, stream>>>(query_key, Qbf);
    copy_qv_kernel <<<(B_ * CV * N_) / 256, 256, 0, stream>>>(query_value, out);

    flash_attn_kernel<<<B_ * (N_ / (WAVES * ROWS)), WAVES * 32, 0, stream>>>(Qbf, Kbf, Vbf, out);
}
